// MambaBlock_3547642986678
// MI455X (gfx1250) — compile-verified, matrix operand order checked by reading, not run on hardware
//
#include <hip/hip_runtime.h>
#include <hip/hip_bf16.h>
#include <stdint.h>

// ---------------------------------------------------------------------------
// Mamba block for MI455X (gfx1250, wave32, WMMA, async-to-LDS staging).
//   0) cvt_f32_bf16       : one-shot bf16 copies of X and weight matrices
//   1) gemm_nt_bf16<64,64>: xz = X @ in_proj_w^T   (global_load_async_to_lds
//                           staging + v_wmma_f32_16x16x32_bf16, 2x2 reg block)
//   2) conv_silu          : xi = silu(causal depthwise conv)   (fp32 + bf16 out)
//   3) gemm_nt_bf16<64,16>: x_dbl = xi @ x_proj_w^T
//   4) scan_fused         : dt_proj+softplus + selective scan + gate (bf16 y out)
//   5) gemm_nt_bf16<64,64>: out_mid = y @ out_proj_w^T
//   6) residual_ln        : out = LayerNorm(out_mid + x)
// ---------------------------------------------------------------------------

typedef __attribute__((ext_vector_type(16))) __bf16 v16bf;
typedef __attribute__((ext_vector_type(8)))  float  v8f;

static constexpr int BATCH   = 8;
static constexpr int SEQ     = 2048;
static constexpr int DMODEL  = 256;
static constexpr int DINNER  = 512;
static constexpr int DSTATE  = 16;
static constexpr int DTRANK  = 16;
static constexpr int NDBL    = DTRANK + 2 * DSTATE;   // 48
static constexpr int MROWS   = BATCH * SEQ;           // 16384
static constexpr int LDK     = 256 + 8;               // LDS row stride (bf16)

// -------- fp32 -> bf16 conversion (one-shot for X and weights) --------------
__global__ __launch_bounds__(256)
void cvt_f32_bf16(const float* __restrict__ in, __bf16* __restrict__ out, int n) {
  int i = blockIdx.x * 256 + threadIdx.x;
  if (i < n) out[i] = (__bf16)in[i];
}

// -------- async stage: ROWS x 256 bf16 tile, global -> LDS ------------------
// 16-byte transfers on the ASYNCcnt path (global_load_async_to_lds_b128).
// LDS dest offset = low 32 bits of the generic shared-pointer (ISA 10.2).
template <int ROWS, int NT>
__device__ __forceinline__ void stage_async(__bf16* __restrict__ s,
                                            const __bf16* __restrict__ g,
                                            int K, int tid) {
  // 32 x 16B chunks per 256-wide row; all addresses 16B aligned (528B stride).
#pragma unroll 4
  for (int idx = tid; idx < ROWS * 32; idx += NT) {
    int r = idx >> 5, c = (idx & 31) << 3;
    unsigned lds = (unsigned)(uintptr_t)(s + r * LDK + c);
    const __bf16* p = g + (size_t)r * K + c;
    asm volatile("global_load_async_to_lds_b128 %0, %1, off"
                 :: "v"(lds), "v"(p) : "memory");
  }
}

__device__ __forceinline__ void wait_async_all() {
  asm volatile("s_wait_asynccnt 0x0" ::: "memory");
}

// -------- WMMA fragment load from LDS (bf16, K-major row, padded stride) ----
// Per-lane layout for 16-bit operands, 16x32 tile:
//   lanes 0-15 : row = lane,    elems 0..7 -> K 0..7,  elems 8..15 -> K 16..23
//   lanes 16-31: row = lane-16, elems 0..7 -> K 8..15, elems 8..15 -> K 24..31
__device__ __forceinline__ v16bf load_frag(const __bf16* __restrict__ p,
                                           int lane, int k0) {
  int r  = lane & 15;
  int kb = k0 + ((lane >> 4) << 3);
  const __bf16* q = p + r * LDK + kb;
  v16bf f;
#pragma unroll
  for (int i = 0; i < 8; ++i) { f[i] = q[i]; f[8 + i] = q[16 + i]; }
  return f;
}

// -------- NT GEMM: C[M,N] = A[M,K] * B[N,K]^T, bf16 in / fp32 out -----------
// Workgroup tile BM x BN; each wave owns a WTM x WTN register tile
// ((WTM/16)x(WTN/16) accumulators) so each fragment is reused across WMMAs.
template <int BM, int BN, int WTM, int WTN>
__global__ __launch_bounds__((BM / WTM) * (BN / WTN) * 32)
void gemm_nt_bf16(const __bf16* __restrict__ A, const __bf16* __restrict__ B,
                  float* __restrict__ C, int M, int N, int K) {
  constexpr int WAVES = (BM / WTM) * (BN / WTN);
  constexpr int NT = WAVES * 32;
  constexpr int TM = WTM / 16, TN = WTN / 16;
  __shared__ __bf16 sA[BM * LDK];
  __shared__ __bf16 sB[BN * LDK];

  const int tid  = threadIdx.x;
  const int wave = tid >> 5, lane = tid & 31;
  const int wm = wave % (BM / WTM), wn = wave / (BM / WTM);
  const int m0 = blockIdx.x * BM, n0 = blockIdx.y * BN;

  v8f acc[TM][TN];
#pragma unroll
  for (int i = 0; i < TM; ++i)
#pragma unroll
    for (int j = 0; j < TN; ++j) acc[i][j] = (v8f){0,0,0,0,0,0,0,0};

  for (int kc = 0; kc < K; kc += 256) {
    stage_async<BM, NT>(sA, A + (size_t)m0 * K + kc, K, tid);
    stage_async<BN, NT>(sB, B + (size_t)n0 * K + kc, K, tid);
    wait_async_all();
    __syncthreads();

    const __bf16* pa = sA + (wm * WTM) * LDK;
    const __bf16* pb = sB + (wn * WTN) * LDK;
#pragma unroll
    for (int kk = 0; kk < 256; kk += 32) {
      v16bf af[TM], bfr[TN];
#pragma unroll
      for (int i = 0; i < TM; ++i) af[i]  = load_frag(pa + 16 * i * LDK, lane, kk);
#pragma unroll
      for (int j = 0; j < TN; ++j) bfr[j] = load_frag(pb + 16 * j * LDK, lane, kk);
#pragma unroll
      for (int i = 0; i < TM; ++i)
#pragma unroll
        for (int j = 0; j < TN; ++j)
          acc[i][j] = __builtin_amdgcn_wmma_f32_16x16x32_bf16(
              false, af[i], false, bfr[j], (short)0, acc[i][j], false, false);
    }
    __syncthreads();
  }

  // C/D layout: VGPR v -> M = v + 8*(lane>=16), N = lane&15
  const int nn = lane & 15;
  const int mb = (lane >> 4) << 3;
#pragma unroll
  for (int i = 0; i < TM; ++i)
#pragma unroll
    for (int j = 0; j < TN; ++j) {
      int mrow = m0 + wm * WTM + 16 * i + mb;
      int ncol = n0 + wn * WTN + 16 * j + nn;
#pragma unroll
      for (int v = 0; v < 8; ++v)
        C[(size_t)(mrow + v) * N + ncol] = acc[i][j][v];
    }
}

// -------- Causal depthwise conv1d (width 4) + bias + SiLU -------------------
__global__ __launch_bounds__(256)
void conv_silu(const float* __restrict__ xz, const float* __restrict__ cw,
               const float* __restrict__ cb, float* __restrict__ xi_f32,
               __bf16* __restrict__ xi_bf) {
  int idx = blockIdx.x * blockDim.x + threadIdx.x;   // over MROWS*DINNER
  int d = idx & (DINNER - 1);
  int m = idx >> 9;
  int l = m & (SEQ - 1);
  const float* col = xz + (size_t)m * (2 * DINNER) + d;   // xi half of xz
  float s = cb[d];
#pragma unroll
  for (int j = 0; j < 4; ++j) {
    int lj = l + j - 3;
    if (lj >= 0) s += cw[d * 4 + j] * col[(long)(j - 3) * (2 * DINNER)];
  }
  float v = s / (1.f + __expf(-s));                       // silu
  xi_f32[idx] = v;
  xi_bf[idx]  = (__bf16)v;                                // GEMM2 operand
}

// -------- Fused dt_proj + softplus + selective scan + gate ------------------
// One block per batch element, one lane per channel d; 16 SSM states in
// registers; per-timestep [dt_rank|B|C] vector (48 floats) staged via LDS.
__global__ __launch_bounds__(DINNER)
void scan_fused(const float* __restrict__ xz, const float* __restrict__ xi_act,
                const float* __restrict__ x_dbl, const float* __restrict__ dtw,
                const float* __restrict__ dtb, const float* __restrict__ A_log,
                const float* __restrict__ Dp, __bf16* __restrict__ y_gated) {
  const int b = blockIdx.x;
  const int d = threadIdx.x;
  __shared__ float sX[NDBL];

  float Ad[DSTATE], h[DSTATE], wdt[DTRANK];
#pragma unroll
  for (int n = 0; n < DSTATE; ++n) {
    Ad[n] = -__expf(A_log[d * DSTATE + n]);    // A = -exp(A_log)
    h[n]  = 0.f;
  }
#pragma unroll
  for (int r = 0; r < DTRANK; ++r) wdt[r] = dtw[d * DTRANK + r];
  const float bias = dtb[d];
  const float Dd   = Dp[d];

  for (int l = 0; l < SEQ; ++l) {
    const int m = b * SEQ + l;
    __syncthreads();                           // previous-step readers done
    if (d < NDBL) sX[d] = x_dbl[(size_t)m * NDBL + d];
    __syncthreads();

    float s = bias;
#pragma unroll
    for (int r = 0; r < DTRANK; ++r) s += wdt[r] * sX[r];
    float dt = (s > 20.f) ? s : log1pf(__expf(s));

    const float xi = xi_act[(size_t)m * DINNER + d];
    const float z  = xz[(size_t)m * (2 * DINNER) + DINNER + d];

    float acc = 0.f;
#pragma unroll
    for (int n = 0; n < DSTATE; ++n) {
      float dA = __expf(dt * Ad[n]);
      h[n] = dA * h[n] + dt * sX[DTRANK + n] * xi;        // h = dA*h + dB*u
      acc += h[n] * sX[DTRANK + DSTATE + n];              // y += h*C
    }
    float y = acc + Dd * xi;
    y_gated[(size_t)m * DINNER + d] = (__bf16)(y * (z / (1.f + __expf(-z))));
  }
}

// -------- Residual add + LayerNorm (one wave per row of 256) ----------------
__global__ __launch_bounds__(256)
void residual_ln(const float* __restrict__ om, const float* __restrict__ x,
                 const float* __restrict__ lw, const float* __restrict__ lb,
                 float* __restrict__ out) {
  const int row  = blockIdx.x * 8 + (threadIdx.x >> 5);
  const int lane = threadIdx.x & 31;
  const float* r0 = om + (size_t)row * DMODEL;
  const float* x0 = x  + (size_t)row * DMODEL;

  float v[8];
  float sum = 0.f;
#pragma unroll
  for (int i = 0; i < 8; ++i) {
    int c = lane + 32 * i;
    v[i] = r0[c] + x0[c];
    sum += v[i];
  }
#pragma unroll
  for (int off = 16; off > 0; off >>= 1) sum += __shfl_xor(sum, off, 32);
  const float mean = sum * (1.f / DMODEL);

  float sq = 0.f;
#pragma unroll
  for (int i = 0; i < 8; ++i) { float t = v[i] - mean; sq += t * t; }
#pragma unroll
  for (int off = 16; off > 0; off >>= 1) sq += __shfl_xor(sq, off, 32);
  const float inv = rsqrtf(sq * (1.f / DMODEL) + 1e-5f);

#pragma unroll
  for (int i = 0; i < 8; ++i) {
    int c = lane + 32 * i;
    out[(size_t)row * DMODEL + c] = (v[i] - mean) * inv * lw[c] + lb[c];
  }
}

// ---------------------------------------------------------------------------
extern "C" void kernel_launch(void* const* d_in, const int* in_sizes, int n_in,
                              void* d_out, int out_size, void* d_ws, size_t ws_size,
                              hipStream_t stream) {
  const float* x          = (const float*)d_in[0];
  const float* in_proj_w  = (const float*)d_in[1];
  const float* conv_w     = (const float*)d_in[2];
  const float* conv_b     = (const float*)d_in[3];
  const float* x_proj_w   = (const float*)d_in[4];
  const float* dt_proj_w  = (const float*)d_in[5];
  const float* dt_proj_b  = (const float*)d_in[6];
  const float* A_log      = (const float*)d_in[7];
  const float* Dvec       = (const float*)d_in[8];
  const float* out_proj_w = (const float*)d_in[9];
  const float* ln_w       = (const float*)d_in[10];
  const float* ln_b       = (const float*)d_in[11];
  float* out = (float*)d_out;

  // workspace layout (all offsets 256B aligned)
  char* ws = (char*)d_ws;
  size_t off = 0;
  auto take = [&](size_t bytes) { char* p = ws + off; off += bytes; return p; };
  float*  xz      = (float*) take((size_t)MROWS * 2 * DINNER * 4);  // 64 MiB
  float*  xi_f32  = (float*) take((size_t)MROWS * DINNER * 4);      // 32 MiB
  __bf16* xi_bf   = (__bf16*)take((size_t)MROWS * DINNER * 2);      // 16 MiB
  float*  x_dbl   = (float*) take((size_t)MROWS * NDBL * 4);        //  3 MiB
  __bf16* y_bf    = (__bf16*)take((size_t)MROWS * DINNER * 2);      // 16 MiB
  float*  out_mid = (float*) take((size_t)MROWS * DMODEL * 4);      // 16 MiB
  __bf16* x_bf    = (__bf16*)take((size_t)MROWS * DMODEL * 2);      //  8 MiB
  __bf16* w1_bf   = (__bf16*)take((size_t)2 * DINNER * DMODEL * 2); // 512 KiB
  __bf16* wx_bf   = (__bf16*)take((size_t)NDBL * DINNER * 2);       //  48 KiB
  __bf16* wo_bf   = (__bf16*)take((size_t)DMODEL * DINNER * 2);     // 256 KiB

  // 0) one-shot bf16 conversions of GEMM operands that start as fp32
  {
    int n;
    n = MROWS * DMODEL;       cvt_f32_bf16<<<(n + 255) / 256, 256, 0, stream>>>(x, x_bf, n);
    n = 2 * DINNER * DMODEL;  cvt_f32_bf16<<<(n + 255) / 256, 256, 0, stream>>>(in_proj_w, w1_bf, n);
    n = NDBL * DINNER;        cvt_f32_bf16<<<(n + 255) / 256, 256, 0, stream>>>(x_proj_w, wx_bf, n);
    n = DMODEL * DINNER;      cvt_f32_bf16<<<(n + 255) / 256, 256, 0, stream>>>(out_proj_w, wo_bf, n);
  }

  // 1) xz = X @ in_proj_w^T        [16384,1024] = [16384,256] x [1024,256]^T
  {
    dim3 g(MROWS / 64, (2 * DINNER) / 64);
    gemm_nt_bf16<64, 64, 32, 32><<<g, 128, 0, stream>>>(x_bf, w1_bf, xz,
                                                        MROWS, 2 * DINNER, DMODEL);
  }
  // 2) xi = silu(conv1d(xi) + b)
  conv_silu<<<(MROWS * DINNER) / 256, 256, 0, stream>>>(xz, conv_w, conv_b,
                                                        xi_f32, xi_bf);
  // 3) x_dbl = xi @ x_proj_w^T     [16384,48] = [16384,512] x [48,512]^T
  {
    dim3 g(MROWS / 64, NDBL / 16);
    gemm_nt_bf16<64, 16, 32, 16><<<g, 64, 0, stream>>>(xi_bf, wx_bf, x_dbl,
                                                       MROWS, NDBL, DINNER);
  }
  // 4) fused dt_proj + softplus + selective scan + D skip + silu(z) gating
  scan_fused<<<BATCH, DINNER, 0, stream>>>(xz, xi_f32, x_dbl, dt_proj_w,
                                           dt_proj_b, A_log, Dvec, y_bf);
  // 5) out_mid = y @ out_proj_w^T  [16384,256] = [16384,512] x [256,512]^T
  {
    dim3 g(MROWS / 64, DMODEL / 64);
    gemm_nt_bf16<64, 64, 32, 32><<<g, 128, 0, stream>>>(y_bf, wo_bf, out_mid,
                                                        MROWS, DMODEL, DINNER);
  }
  // 6) out = LayerNorm(out_mid + x)
  residual_ln<<<MROWS / 8, 256, 0, stream>>>(out_mid, x, ln_w, ln_b, out);
}
